// STGCN_62749472195368
// MI455X (gfx1250) — matrix operand order repaired and checked by reading, not compiled, not run on hardware
//
#include <hip/hip_runtime.h>

// ---------------------------------------------------------------------------
// STGCN forward for MI455X (gfx1250, wave32).
// Heavy TCN convs run as bf16 WMMA GEMMs (v_wmma_f32_16x16x32_bf16) with
// fused bias/BN/ReLU/residual epilogues; graph attention stays in f32.
// Weights are pre-converted to bf16 in WMMA-friendly layout; each conv block
// processes 8 rows to amortize LDS weight staging (keeps weight traffic L2-
// resident; activation traffic ~1 GB bf16 => ~45us at 23.3 TB/s is the bound).
// ---------------------------------------------------------------------------

typedef __attribute__((ext_vector_type(16))) __bf16 v16bf;
typedef __attribute__((ext_vector_type(8)))  float  v8f;

#define Bd   64
#define Nn   150
#define Fin  10
#define Tt   64
#define Dd   64
#define Kk   30
#define Mtot (Bd * Tt * Nn)      // 614400 nodes
#define Rtot (Bd * Nn)           // 9600 TCN rows

__device__ __forceinline__ v16bf load16(const __bf16* p0, const __bf16* p1) {
    v16bf r;
#pragma unroll
    for (int i = 0; i < 8; ++i) { r[i] = p0[i]; r[i + 8] = p1[i]; }
    return r;
}

// ---------------------------------------------------------------------------
// 1. embedding row norms
// ---------------------------------------------------------------------------
__global__ void norm_kernel(const float* __restrict__ emb, float* __restrict__ nrm) {
    int i = blockIdx.x * blockDim.x + threadIdx.x;
    if (i < Nn) {
        float s = 0.f;
        for (int d = 0; d < Dd; ++d) { float e = emb[i * Dd + d]; s += e * e; }
        nrm[i] = sqrtf(s);
    }
}

// ---------------------------------------------------------------------------
// 2. cosine top-K per row (one wave per row; iterative argmax, K=30)
// ---------------------------------------------------------------------------
__global__ __launch_bounds__(32)
void topk_kernel(const float* __restrict__ emb, const float* __restrict__ nrm,
                 int* __restrict__ idx) {
    __shared__ float cs[Nn + 2];
    const int i = blockIdx.x, lane = threadIdx.x;
    for (int j = lane; j < Nn; j += 32) {
        float d = 0.f;
        for (int k = 0; k < Dd; ++k) d += emb[i * Dd + k] * emb[j * Dd + k];
        cs[j] = d / (nrm[i] * nrm[j]);
    }
    __syncthreads();
    for (int k = 0; k < Kk; ++k) {
        float best = -1e30f; int bi = 1 << 20;
        for (int j = lane; j < Nn; j += 32)
            if (cs[j] > best) { best = cs[j]; bi = j; }
        for (int off = 16; off; off >>= 1) {
            float ob = __shfl_xor(best, off, 32);
            int   oi = __shfl_xor(bi,   off, 32);
            if (ob > best || (ob == best && oi < bi)) { best = ob; bi = oi; }
        }
        if (lane == 0) { idx[i * Kk + k] = bi; cs[bi] = -1e30f; }
        __syncthreads();
    }
}

// ---------------------------------------------------------------------------
// 3. lift: xl = x @ lin_w^T ; a_i, a_j ; default h0 = bf16(xl + gnn_bias)
//    x[m][f] = data[((b*N+n)*F + f)*T + t],  m = (b*T+t)*N + n
//    h0 layout: [r = b*N+n][d][t]  (bf16)
// ---------------------------------------------------------------------------
__global__ __launch_bounds__(256)
void xlift_kernel(const float* __restrict__ data, const float* __restrict__ lin_w,
                  const float* __restrict__ att_i, const float* __restrict__ att_j,
                  const float* __restrict__ att_em_i, const float* __restrict__ att_em_j,
                  const float* __restrict__ emb, const float* __restrict__ gnn_bias,
                  float* __restrict__ xl, float* __restrict__ ai, float* __restrict__ aj,
                  __bf16* __restrict__ h0) {
    __shared__ float lw[Dd * Fin];
    __shared__ float Ai[Dd], Aj[Dd], Aei[Dd], Aej[Dd], Gb[Dd];
    const int tid = threadIdx.x;
    for (int i = tid; i < Dd * Fin; i += 256) lw[i] = lin_w[i];
    if (tid < Dd) {
        Ai[tid] = att_i[tid]; Aj[tid] = att_j[tid];
        Aei[tid] = att_em_i[tid]; Aej[tid] = att_em_j[tid];
        Gb[tid] = gnn_bias[tid];
    }
    __syncthreads();

    const int m = blockIdx.x * 256 + tid;              // Mtot % 256 == 0
    const int b = m / (Tt * Nn);
    const int rem = m - b * Tt * Nn;
    const int t = rem / Nn;
    const int n = rem - t * Nn;
    const int r = b * Nn + n;

    float xr[Fin];
#pragma unroll
    for (int f = 0; f < Fin; ++f)
        xr[f] = data[((size_t)(b * Nn + n) * Fin + f) * Tt + t];

    const float* er = emb + n * Dd;
    float sai = 0.f, saj = 0.f;
#pragma unroll 4
    for (int d = 0; d < Dd; ++d) {
        float s = 0.f;
#pragma unroll
        for (int f = 0; f < Fin; ++f) s += xr[f] * lw[d * Fin + f];
        xl[(size_t)m * Dd + d] = s;
        sai += s * Ai[d] + er[d] * Aei[d];
        saj += s * Aj[d] + er[d] * Aej[d];
        h0[((size_t)r * Dd + d) * Tt + t] = (__bf16)(s + Gb[d]);
    }
    ai[m] = sai; aj[m] = saj;
}

// ---------------------------------------------------------------------------
// 4. attention aggregation for first GB=64 graphs (b=0, all t): wave per node
// ---------------------------------------------------------------------------
__global__ __launch_bounds__(32)
void attn_kernel(const int* __restrict__ idx, const float* __restrict__ ai,
                 const float* __restrict__ aj, const float* __restrict__ xl,
                 const float* __restrict__ gb, __bf16* __restrict__ h0) {
    const int n = blockIdx.x;        // node in graph, 0..149
    const int g = blockIdx.y;        // graph 0..63  (== t, since b=0)
    const int lane = threadIdx.x;
    const int mdst = g * Nn + n;

    int msrc = mdst;
    float alpha = -1e30f;
    if (lane < Kk) {
        int j = idx[n * Kk + lane];
        msrc = g * Nn + j;
        float a = ai[mdst] + aj[msrc];
        alpha = a > 0.f ? a : 0.2f * a;           // leaky_relu(0.2)
    }
    float amax = alpha;
    for (int off = 16; off; off >>= 1) amax = fmaxf(amax, __shfl_xor(amax, off, 32));
    float ex = (lane < Kk) ? expf(alpha - amax) : 0.f;
    float den = ex;
    for (int off = 16; off; off >>= 1) den += __shfl_xor(den, off, 32);
    const float w = ex / den;

    float o0 = 0.f, o1 = 0.f;
    for (int k = 0; k < Kk; ++k) {
        float wk = __shfl(w, k, 32);
        int   ms = __shfl(msrc, k, 32);
        o0 += wk * xl[(size_t)ms * Dd + lane];
        o1 += wk * xl[(size_t)ms * Dd + 32 + lane];
    }
    // m = g*N + n  ->  b=0, t=g, n  ->  r = n
    h0[((size_t)n * Dd + lane) * Tt + g]      = (__bf16)(o0 + gb[lane]);
    h0[((size_t)n * Dd + lane + 32) * Tt + g] = (__bf16)(o1 + gb[lane + 32]);
}

// ---------------------------------------------------------------------------
// 5. depthwise conv k3, pad 1  (cheap, VALU)
// ---------------------------------------------------------------------------
__global__ __launch_bounds__(256)
void dw_kernel(const __bf16* __restrict__ X, const float* __restrict__ w,
               const float* __restrict__ b, __bf16* __restrict__ Y, int total) {
    int i = blockIdx.x * 256 + threadIdx.x;
    if (i >= total) return;
    const int t = i & 63;
    const int c = (i >> 6) & 63;
    float acc = b[c];
#pragma unroll
    for (int dt = 0; dt < 3; ++dt) {
        int tt = t + dt - 1;
        if (tt >= 0 && tt < Tt) acc += (float)X[i + (tt - t)] * w[c * 3 + dt];
    }
    Y[i] = (__bf16)acc;
}

// ---------------------------------------------------------------------------
// 6a. weight pre-convert: f32 [cout][cin][kw] -> bf16 [dt][cout][cin]
//     (done once per conv launch; conv blocks then stage with b128 copies)
// ---------------------------------------------------------------------------
__global__ __launch_bounds__(256)
void wcvt_kernel(const float* __restrict__ W, __bf16* __restrict__ Wb,
                 int cinv, int kwv, int total) {
    int i = blockIdx.x * 256 + threadIdx.x;
    if (i >= total) return;
    int dt   = i / (Dd * cinv);
    int rem  = i - dt * Dd * cinv;
    int cout = rem / cinv;
    int cin  = rem - cout * cinv;
    Wb[i] = (__bf16)W[((size_t)cout * cinv + cin) * kwv + dt];
}

// ---------------------------------------------------------------------------
// 6b. WMMA conv1d: O[64,T] = sum_dt W_dt[64,CIN] @ X[CIN, T(+dt-pad)] + epi
//    EPI 0: pw   -> y = (acc+b)*bn_s + bn_b                  (res path)
//    EPI 1: c3/c5-> y = relu((acc+b)*bn_s + bn_b)            (concat halves)
//    EPI 2: fus  -> y = relu(acc + b + res[r,c,t])           (block output)
//    One block (4 waves) handles ROWS rows; weights staged once, A fragments
//    hoisted to registers and reused across all 4 T-tiles of every row.
// ---------------------------------------------------------------------------
template <int CIN, int KW, int EPI, int ROWS>
__global__ __launch_bounds__(128)
void conv_wmma_kernel(const __bf16* __restrict__ X, const __bf16* __restrict__ Wb,
                      const float* __restrict__ bias, const float* __restrict__ bng,
                      const float* __restrict__ bnb, const __bf16* __restrict__ res,
                      __bf16* __restrict__ Y, int yldc, int yco) {
    constexpr int PAD  = (KW - 1) / 2;
    constexpr int TP   = Tt + 2 * PAD;
    constexpr int CINP = CIN + 8;               // row pad: keeps 16B alignment, kills conflicts
    constexpr int KC   = CIN / 32;

    __shared__ __align__(16) __bf16 Xs[TP * CINP];        // Xs[tp][cin]
    __shared__ __align__(16) __bf16 Ws[KW * Dd * CIN];    // Ws[dt][cout][cin]
    __shared__ float Bs[Dd], Gs[Dd], Hs[Dd];

    const int tid = threadIdx.x;

    // stage bf16 weights with 16B copies (layout already matches)
    constexpr int WV = KW * Dd * CIN / 8;       // # of uint4 chunks
    for (int i = tid; i < WV; i += 128)
        ((uint4*)Ws)[i] = ((const uint4*)Wb)[i];
    if (tid < Dd) {
        Bs[tid] = bias[tid];
        if (EPI != 2) { Gs[tid] = bng[tid] * rsqrtf(1.0f + 1e-5f); Hs[tid] = bnb[tid]; }
    }

    const int lane = tid & 31;
    const int wave = tid >> 5;
    const int lm   = lane & 15;     // A: M row / B,D: N column
    const int lh   = lane >> 4;     // K-half select
    const int m0   = wave * 16;     // Cout tile

    for (int rr = 0; rr < ROWS; ++rr) {
        const int r = blockIdx.x * ROWS + rr;
        __syncthreads();            // weights ready / previous row's reads done
        for (int i = tid; i < CIN * TP; i += 128) {
            int cin = i / TP;
            int tp  = i - cin * TP;
            int t   = tp - PAD;
            float v = (t >= 0 && t < Tt) ? (float)X[((size_t)r * CIN + cin) * Tt + t] : 0.f;
            Xs[tp * CINP + cin] = (__bf16)v;
        }
        __syncthreads();

        // hoist A fragments (invariant across T-tiles)
        v16bf afr[KW * KC];
#pragma unroll
        for (int dt = 0; dt < KW; ++dt)
#pragma unroll
            for (int kc = 0; kc < KC; ++kc) {
                // A frag (16x32 bf16): lane row cout=m0+lm; elems 0..7 -> K=lh*8..,
                // elems 8..15 -> K=16+lh*8..  (ISA 16-bit A layout)
                const __bf16* ap = &Ws[(dt * Dd + m0 + lm) * CIN + kc * 32 + lh * 8];
                afr[dt * KC + kc] = load16(ap, ap + 16);
            }

#pragma unroll
        for (int n0 = 0; n0 < Tt; n0 += 16) {
            v8f acc = {0.f, 0.f, 0.f, 0.f, 0.f, 0.f, 0.f, 0.f};
#pragma unroll
            for (int dt = 0; dt < KW; ++dt) {
#pragma unroll
                for (int kc = 0; kc < KC; ++kc) {
                    // B frag (32x16 bf16): lane column t=n0+lm; elems 0..7 -> K=lh*8.., elems 8..15 -> K=16+lh*8..
                    const __bf16* bp = &Xs[(n0 + lm + dt) * CINP + kc * 32 + lh * 8];
                    v16bf b = load16(bp, bp + 16);
                    acc = __builtin_amdgcn_wmma_f32_16x16x32_bf16(
                              false, afr[dt * KC + kc], false, b, (short)0, acc,
                              false, false);
                }
            }
#pragma unroll
            for (int v = 0; v < 8; ++v) {       // D: m = v + 8*lh, n = lm
                const int c = m0 + v + lh * 8;
                const int t = n0 + lm;
                float y = acc[v];
                if (EPI == 0) {
                    y = (y + Bs[c]) * Gs[c] + Hs[c];
                } else if (EPI == 1) {
                    y = (y + Bs[c]) * Gs[c] + Hs[c];
                    y = y > 0.f ? y : 0.f;
                } else {
                    y = y + Bs[c] + (float)res[((size_t)r * Dd + c) * Tt + t];
                    y = y > 0.f ? y : 0.f;
                }
                Y[((size_t)r * yldc + yco + c) * Tt + t] = (__bf16)y;
            }
        }
    }
}

// ---------------------------------------------------------------------------
// 7. head: mean over T, *emb, BN, ReLU, dot out_w  -> out[B*N]
// ---------------------------------------------------------------------------
__global__ __launch_bounds__(32)
void head_kernel(const __bf16* __restrict__ h, const float* __restrict__ emb,
                 const float* __restrict__ g, const float* __restrict__ b,
                 const float* __restrict__ ow, const float* __restrict__ ob,
                 float* __restrict__ out) {
    const int r = blockIdx.x;
    const int n = r % Nn;
    const int lane = threadIdx.x;
    const __bf16* p0 = h + ((size_t)r * Dd + lane) * Tt;
    const __bf16* p1 = h + ((size_t)r * Dd + lane + 32) * Tt;
    float s0 = 0.f, s1 = 0.f;
    for (int t = 0; t < Tt; ++t) { s0 += (float)p0[t]; s1 += (float)p1[t]; }
    s0 *= (1.f / 64.f); s1 *= (1.f / 64.f);
    const float sc = rsqrtf(1.f + 1e-5f);
    float v0 = s0 * emb[n * Dd + lane]      * (g[lane] * sc)      + b[lane];
    float v1 = s1 * emb[n * Dd + lane + 32] * (g[lane + 32] * sc) + b[lane + 32];
    v0 = v0 > 0.f ? v0 : 0.f;
    v1 = v1 > 0.f ? v1 : 0.f;
    float p = v0 * ow[lane] + v1 * ow[lane + 32];
    for (int off = 16; off; off >>= 1) p += __shfl_xor(p, off, 32);
    if (lane == 0) out[r] = p + ob[0];
}

// ---------------------------------------------------------------------------
// launch
// ---------------------------------------------------------------------------
struct BlkPtr {
    const float *c3w, *c3b, *bn3g, *bn3b, *c5w, *c5b, *bn5g, *bn5b;
    const float *dww, *dwb, *pww, *pwb, *rbg, *rbb, *fw, *fb;
};

extern "C" void kernel_launch(void* const* d_in, const int* in_sizes, int n_in,
                              void* d_out, int out_size, void* d_ws, size_t ws_size,
                              hipStream_t stream) {
    (void)in_sizes; (void)n_in; (void)out_size; (void)ws_size;
    // ---- inputs (recursive insertion order of setup_inputs dict) ----
    const float* data = (const float*)d_in[0];
    const float* emb  = (const float*)d_in[1];
    const float* lin_w = (const float*)d_in[2];
    const float* att_i = (const float*)d_in[3];
    const float* att_j = (const float*)d_in[4];
    const float* att_em_i = (const float*)d_in[5];
    const float* att_em_j = (const float*)d_in[6];
    const float* gnn_bias = (const float*)d_in[7];
    auto blk = [&](int base) {
        BlkPtr p;
        p.c3w = (const float*)d_in[base + 0];  p.c3b = (const float*)d_in[base + 1];
        p.bn3g = (const float*)d_in[base + 2]; p.bn3b = (const float*)d_in[base + 3];
        p.c5w = (const float*)d_in[base + 4];  p.c5b = (const float*)d_in[base + 5];
        p.bn5g = (const float*)d_in[base + 6]; p.bn5b = (const float*)d_in[base + 7];
        p.dww = (const float*)d_in[base + 8];  p.dwb = (const float*)d_in[base + 9];
        p.pww = (const float*)d_in[base + 10]; p.pwb = (const float*)d_in[base + 11];
        p.rbg = (const float*)d_in[base + 12]; p.rbb = (const float*)d_in[base + 13];
        p.fw  = (const float*)d_in[base + 14]; p.fb  = (const float*)d_in[base + 15];
        return p;
    };
    const BlkPtr b1 = blk(8), b2 = blk(24);
    const float* bn_out_g = (const float*)d_in[40];
    const float* bn_out_b = (const float*)d_in[41];
    const float* out_w    = (const float*)d_in[42];
    const float* out_b    = (const float*)d_in[43];
    float* out = (float*)d_out;

    // ---- workspace carve-up ----
    char* ws = (char*)d_ws;
    size_t off = 0;
    auto carve = [&](size_t bytes) {
        void* p = (void*)(ws + off);
        off = (off + bytes + 255) & ~(size_t)255;
        return p;
    };
    int*    idx  = (int*)carve((size_t)Nn * Kk * 4);
    float*  nrm  = (float*)carve((size_t)Nn * 4);
    float*  ai   = (float*)carve((size_t)Mtot * 4);
    float*  aj   = (float*)carve((size_t)Mtot * 4);
    float*  xl   = (float*)carve((size_t)Mtot * Dd * 4);
    __bf16* hA   = (__bf16*)carve((size_t)Rtot * Dd * Tt * 2);
    __bf16* dwt  = (__bf16*)carve((size_t)Rtot * Dd * Tt * 2);
    __bf16* resb = (__bf16*)carve((size_t)Rtot * Dd * Tt * 2);
    __bf16* cat  = (__bf16*)carve((size_t)Rtot * 2 * Dd * Tt * 2);
    __bf16* hB   = (__bf16*)carve((size_t)Rtot * Dd * Tt * 2);
    __bf16* wbuf = (__bf16*)carve((size_t)5 * Dd * 128 * 2);   // max conv weight (c5)

    // ---- graph ----
    norm_kernel<<<1, 256, 0, stream>>>(emb, nrm);
    topk_kernel<<<Nn, 32, 0, stream>>>(emb, nrm, idx);

    // ---- lift + default gcn, attention overwrite for first GB graphs ----
    xlift_kernel<<<Mtot / 256, 256, 0, stream>>>(
        data, lin_w, att_i, att_j, att_em_i, att_em_j, emb, gnn_bias,
        xl, ai, aj, hA);
    attn_kernel<<<dim3(Nn, Bd), 32, 0, stream>>>(idx, ai, aj, xl, gnn_bias, hA);

    // ---- multiscale TCN blocks (bf16 WMMA) ----
    constexpr int ROWS = 8;                    // 9600 / 8 = 1200 blocks
    const int dwTot = Rtot * Dd * Tt;
    auto cvt = [&](const float* W, int cin, int kw) {
        int total = kw * Dd * cin;
        wcvt_kernel<<<(total + 255) / 256, 256, 0, stream>>>(W, wbuf, cin, kw, total);
    };
    auto run_block = [&](const __bf16* hin, __bf16* hout, const BlkPtr& p) {
        dw_kernel<<<(dwTot + 255) / 256, 256, 0, stream>>>(hin, p.dww, p.dwb, dwt, dwTot);
        cvt(p.pww, 64, 1);
        conv_wmma_kernel<64, 1, 0, ROWS><<<Rtot / ROWS, 128, 0, stream>>>(
            dwt, wbuf, p.pwb, p.rbg, p.rbb, nullptr, resb, 64, 0);
        cvt(p.c3w, 64, 3);
        conv_wmma_kernel<64, 3, 1, ROWS><<<Rtot / ROWS, 128, 0, stream>>>(
            hin, wbuf, p.c3b, p.bn3g, p.bn3b, nullptr, cat, 128, 0);
        cvt(p.c5w, 64, 5);
        conv_wmma_kernel<64, 5, 1, ROWS><<<Rtot / ROWS, 128, 0, stream>>>(
            hin, wbuf, p.c5b, p.bn5g, p.bn5b, nullptr, cat, 128, 64);
        cvt(p.fw, 128, 1);
        conv_wmma_kernel<128, 1, 2, ROWS><<<Rtot / ROWS, 128, 0, stream>>>(
            cat, wbuf, p.fb, nullptr, nullptr, resb, hout, 64, 0);
    };
    run_block(hA, hB, b1);
    run_block(hB, hA, b2);

    // ---- head ----
    head_kernel<<<Rtot, 32, 0, stream>>>(hA, emb, bn_out_g, bn_out_b, out_w, out_b, out);
}
